// LNGRU_24515673325922
// MI455X (gfx1250) — compile-verified, matrix operand order checked by reading, not run on hardware
//
#include <hip/hip_runtime.h>
#include <hip/hip_bf16.h>

// ---------------- constants ----------------
constexpr int kT   = 512;
constexpr int kB   = 32;
constexpr int kIN  = 512;
constexpr int kH   = 512;
constexpr int kH3  = 1536;
constexpr int kTB  = kT * kB;          // 16384 rows
constexpr float kEPS = 1e-5f;

// ---------------- types ----------------
typedef __attribute__((ext_vector_type(16))) __bf16 bf16x16;
typedef __attribute__((ext_vector_type(8)))  float  floatx8;

struct U4 { unsigned int a, b, c, d; };
union Frag16 { bf16x16 v; U4 q[2]; };

static __device__ inline unsigned short f2bf(float f) {
    unsigned int u = __float_as_uint(f);
    unsigned int r = (u + 0x7FFFu + ((u >> 16) & 1u)) >> 16;
    return (unsigned short)r;
}
static __device__ inline float bf2f(unsigned short h) {
    return __uint_as_float(((unsigned int)h) << 16);
}
static __device__ inline float sigf(float x) { return 1.0f / (1.0f + __expf(-x)); }

static __device__ inline floatx8 zero8() {
    floatx8 z = {0.f,0.f,0.f,0.f,0.f,0.f,0.f,0.f};
    return z;
}

// Load a 16x32 bf16 WMMA A/B fragment per ISA 7.12.2 layout.
// p points at (row, kbase + half*8) in a row-major bf16 matrix (or LDS tile).
static __device__ inline void load_frag(Frag16& f, const unsigned short* p) {
    f.q[0] = *(const U4*)(p);        // K = kb + half*8 .. +7
    f.q[1] = *(const U4*)(p + 16);   // K = kb + 16 + half*8 .. +7
}

// ---------------- kernel 1: weight convert + transpose ----------------
// WxT[dl][n][k] = (bf16) Wx[dl][k][n]; same for Wh.
__global__ __launch_bounds__(256) void k_cvt_weights(
    const float* __restrict__ Wx, unsigned short* __restrict__ WxT,
    const float* __restrict__ Wh, unsigned short* __restrict__ WhT)
{
    int e = blockIdx.x * 256 + threadIdx.x;                 // over 4*1536*512
    if (e >= 4 * kH3 * kIN) return;
    int dl = e / (kH3 * kIN);
    int r  = e % (kH3 * kIN);
    int n  = r / kIN;
    int k  = r % kIN;
    size_t src = (size_t)dl * kIN * kH3 + (size_t)k * kH3 + n;
    WxT[e] = f2bf(Wx[src]);
    WhT[e] = f2bf(Wh[src]);
}

// ---------------- kernel 2: f32 -> bf16 ----------------
__global__ __launch_bounds__(256) void k_cvt_x(
    const float* __restrict__ src, unsigned short* __restrict__ dst, int n)
{
    int e = blockIdx.x * 256 + threadIdx.x;
    if (e < n) dst[e] = f2bf(src[e]);
}

// ---------------- kernel 3: Ax = Xbf @ WxT^T + bx (raw, bf16 out) ----------------
// grid (24, 128, 2), block 256 (8 waves). Block tile: M=128, N=64; wave: 16x64.
// B tile (shared by all 8 waves) staged through double-buffered LDS; next tile's
// global load is issued before the current tile's WMMAs (2-stage pipeline).
__global__ __launch_bounds__(256) void k_gemm_ax(
    const unsigned short* __restrict__ A0, const unsigned short* __restrict__ A1,
    const unsigned short* __restrict__ WxT, const float* __restrict__ bx,
    unsigned short* __restrict__ Ax, int layer)
{
    __shared__ unsigned short sB[2][64 * 32];   // 2 x 4KB

    int dir = blockIdx.z;
    int dl  = layer * 2 + dir;
    const unsigned short* A  = dir ? A1 : A0;
    const unsigned short* Bw = WxT + (size_t)dl * kH3 * kIN;
    unsigned short* out      = Ax + (size_t)dir * kTB * kH3;
    const float* bxr         = bx + (size_t)dl * kH3;

    int tid  = threadIdx.x;
    int lane = tid & 31;
    int wave = tid >> 5;
    int half = lane >> 4;
    int l15  = lane & 15;

    int m0 = blockIdx.y * 128 + wave * 16;   // M base of this wave
    int nb = blockIdx.x * 64;                // N base of this block

    // cooperative B-tile loader: thread -> (row, 8-elem chunk); 256 thr x 16B = 4KB
    int lrow   = tid >> 2;          // 0..63
    int lchunk = (tid & 3) * 8;     // 0,8,16,24
    const unsigned short* bsrc = Bw + (size_t)(nb + lrow) * kIN + lchunk;

    floatx8 acc[4];
    #pragma unroll
    for (int i = 0; i < 4; ++i) acc[i] = zero8();

    // preload tile kk=0
    *(U4*)(&sB[0][lrow * 32 + lchunk]) = *(const U4*)(bsrc);
    __syncthreads();

    const unsigned short* arow = A + (size_t)(m0 + l15) * kIN;

    for (int kk = 0; kk < 16; ++kk) {
        int kb  = kk * 32;
        int cur = kk & 1;
        int nxt = cur ^ 1;

        // issue next tile's global load early (hidden behind WMMAs)
        U4 nx = {0u, 0u, 0u, 0u};
        if (kk < 15) nx = *(const U4*)(bsrc + kb + 32);

        Frag16 af;
        load_frag(af, arow + kb + half * 8);

        Frag16 bfr[4];
        #pragma unroll
        for (int i = 0; i < 4; ++i)
            load_frag(bfr[i], &sB[cur][(i * 16 + l15) * 32 + half * 8]);

        #pragma unroll
        for (int i = 0; i < 4; ++i)
            acc[i] = __builtin_amdgcn_wmma_f32_16x16x32_bf16(
                false, af.v, false, bfr[i].v, (short)0, acc[i], false, false);

        if (kk < 15) *(U4*)(&sB[nxt][lrow * 32 + lchunk]) = nx;
        __syncthreads();
    }

    #pragma unroll
    for (int i = 0; i < 4; ++i) {
        int n = nb + i * 16 + l15;
        float bv = bxr[n];
        #pragma unroll
        for (int r = 0; r < 8; ++r) {
            int row = m0 + 8 * half + r;
            out[(size_t)row * kH3 + n] = f2bf(acc[i][r] + bv);
        }
    }
}

// ---------------- kernel 4: in-place LN over each (row, gate) 512-chunk ----------------
// One wave per task; tasks = 2 dirs * 16384 rows * 3 gates = 98304.
__global__ __launch_bounds__(256) void k_ln_ax(
    unsigned short* __restrict__ Ax, const float* __restrict__ gx,
    const float* __restrict__ bex, int layer)
{
    int tid  = threadIdx.x;
    int lane = tid & 31;
    int wave = tid >> 5;
    int task = blockIdx.x * 8 + wave;
    int dir  = task / (kTB * 3);
    int rem  = task % (kTB * 3);
    int row  = rem / 3;
    int g    = rem % 3;
    int dl   = layer * 2 + dir;

    unsigned short* p = Ax + (size_t)dir * kTB * kH3 + (size_t)row * kH3 + g * kH;
    int j0 = lane * 16;

    float v[16];
    {
        // exactly this lane's 16 bf16 = 32 bytes = two b128 loads
        const U4* q = (const U4*)(p + j0);
        U4 a = q[0], b = q[1];
        unsigned int w[8] = {a.a, a.b, a.c, a.d, b.a, b.b, b.c, b.d};
        #pragma unroll
        for (int i = 0; i < 8; ++i) {
            v[2 * i]     = bf2f((unsigned short)(w[i] & 0xffff));
            v[2 * i + 1] = bf2f((unsigned short)(w[i] >> 16));
        }
    }

    float s1 = 0.f, s2 = 0.f;
    #pragma unroll
    for (int i = 0; i < 16; ++i) { s1 += v[i]; s2 += v[i] * v[i]; }
    for (int m = 16; m > 0; m >>= 1) {
        s1 += __shfl_xor(s1, m, 32);
        s2 += __shfl_xor(s2, m, 32);
    }
    float mu  = s1 * (1.0f / kH);
    float var = s2 * (1.0f / kH) - mu * mu;
    float rs  = rsqrtf(var + kEPS);

    const float* gxr  = gx  + (size_t)dl * kH3 + g * kH;
    const float* bexr = bex + (size_t)dl * kH3 + g * kH;
    #pragma unroll
    for (int i = 0; i < 16; ++i) {
        int j = j0 + i;
        p[j] = f2bf((v[i] - mu) * rs * gxr[j] + bexr[j]);
    }
}

// ---------------- kernel 5: recurrent scan ----------------
// grid = 2 (direction), block = 1024 (32 waves, one full WGP).
// LDS layout (dynamic, 314112 bytes):
//   sAH  f32 [32][1536]     196608 B  @ 0
//   sHB  bf16[32][512]       32768 B  @ 196608
//   sHF  f32 [32][512]       65536 B  @ 229376
//   sGH  f32 [1536]           6144 B  @ 294912
//   sBEH f32 [1536]           6144 B  @ 301056
//   sBH  f32 [1536]           6144 B  @ 307200
//   sMU  f32 [96]              384 B  @ 313344
//   sRS  f32 [96]              384 B  @ 313728
__global__ __launch_bounds__(1024, 1) void k_gru(
    int layer,
    const unsigned short* __restrict__ Ax,   // post-LN ax, bf16, [dir][row][1536]
    const unsigned short* __restrict__ WhT,  // [dl][n][k] bf16
    const float* __restrict__ bh,
    const float* __restrict__ gh,
    const float* __restrict__ beh,
    const float* __restrict__ h0,
    unsigned short* __restrict__ outFFbf,    // layer 0, dir 0 output (bf16)
    unsigned short* __restrict__ outBKbf,    // layer 0, dir 1 output (bf16)
    float* __restrict__ outF,                // layer 1 output (f32, [t][b][1024])
    float* __restrict__ outHid)              // [4][32][512] f32
{
    extern __shared__ char smem[];
    float*          sAH  = (float*)(smem);
    unsigned short* sHB  = (unsigned short*)(smem + 196608);
    float*          sHF  = (float*)(smem + 229376);
    float*          sGH  = (float*)(smem + 294912);
    float*          sBEH = (float*)(smem + 301056);
    float*          sBH  = (float*)(smem + 307200);
    float*          sMU  = (float*)(smem + 313344);
    float*          sRS  = (float*)(smem + 313728);

    int dir = blockIdx.x;
    int dl  = layer * 2 + dir;
    const unsigned short* Wh_d = WhT + (size_t)dl * kH3 * kIN;
    const unsigned short* Ax_d = Ax  + (size_t)dir * kTB * kH3;
    unsigned short* outbf = dir ? outBKbf : outFFbf;

    int tid  = threadIdx.x;
    int lane = tid & 31;
    int wave = tid >> 5;
    int half = lane >> 4;
    int l15  = lane & 15;

    // ---- init: h, gh/beh/bh into LDS ----
    for (int e = tid; e < kH3; e += 1024) {
        sGH[e]  = gh[(size_t)dl * kH3 + e];
        sBEH[e] = beh[(size_t)dl * kH3 + e];
        sBH[e]  = bh[(size_t)dl * kH3 + e];
    }
    for (int e = tid; e < kB * kH; e += 1024) {
        float v = h0[(size_t)dl * kB * kH + e];
        sHF[e] = v;
        sHB[e] = f2bf(v);
    }
    __syncthreads();

    // wave tiling for the 32x1536 GEMM: 2 M-tiles x 96 N-tiles, 6 tiles/wave
    int mt  = wave >> 4;        // 0..1
    int ntb = wave & 15;        // N tile i: ntb + 16*i, i=0..5
    const unsigned short* hrow = sHB + (size_t)(mt * 16 + l15) * kH;

    // phase-3 assignment: thread -> (batch b, 16 consecutive j)
    int pb  = tid >> 5;
    int pj0 = (tid & 31) * 16;

    for (int s = 0; s < kT; ++s) {
        // ---------- phase 1: ah_raw = h @ Wh + bh -> LDS ----------
        floatx8 acc[6];
        #pragma unroll
        for (int i = 0; i < 6; ++i) acc[i] = zero8();

        for (int kk = 0; kk < 16; ++kk) {
            int kb = kk * 32;

            // batch all loads for this K-chunk first: 12 global b128 + 2 ds b128,
            // so WMMAs can start on partial LOADcnt instead of draining per tile.
            Frag16 bfr[6];
            #pragma unroll
            for (int i = 0; i < 6; ++i) {
                int n = ntb * 16 + 256 * i + l15;
                load_frag(bfr[i], Wh_d + (size_t)n * kIN + kb + half * 8);
            }
            Frag16 af;
            load_frag(af, hrow + kb + half * 8);
            __builtin_prefetch(Wh_d + (size_t)(ntb * 16 + l15) * kIN + kb + 32, 0, 1);

            #pragma unroll
            for (int i = 0; i < 6; ++i)
                acc[i] = __builtin_amdgcn_wmma_f32_16x16x32_bf16(
                    false, af.v, false, bfr[i].v, (short)0, acc[i], false, false);
        }
        #pragma unroll
        for (int i = 0; i < 6; ++i) {
            int n = ntb * 16 + 256 * i + l15;
            float bv = sBH[n];
            #pragma unroll
            for (int r = 0; r < 8; ++r) {
                int b = mt * 16 + 8 * half + r;
                sAH[(size_t)b * kH3 + n] = acc[i][r] + bv;
            }
        }
        __syncthreads();

        // ---------- phase 2: LN stats per (b, gate) ----------
        #pragma unroll
        for (int i = 0; i < 3; ++i) {
            int task = wave + 32 * i;       // 0..95
            int b = task / 3, g = task % 3;
            const float* p = sAH + (size_t)b * kH3 + g * kH + lane * 16;
            float s1 = 0.f, s2 = 0.f;
            #pragma unroll
            for (int e = 0; e < 16; ++e) { float v = p[e]; s1 += v; s2 += v * v; }
            for (int m = 16; m > 0; m >>= 1) {
                s1 += __shfl_xor(s1, m, 32);
                s2 += __shfl_xor(s2, m, 32);
            }
            if (lane == 0) {
                float mu  = s1 * (1.0f / kH);
                float var = s2 * (1.0f / kH) - mu * mu;
                sMU[task] = mu;
                sRS[task] = rsqrtf(var + kEPS);
            }
        }
        __syncthreads();

        // ---------- phase 3: gates + state update ----------
        int t = dir ? (kT - 1 - s) : s;
        {
            float mu0 = sMU[pb * 3 + 0], rs0 = sRS[pb * 3 + 0];
            float mu1 = sMU[pb * 3 + 1], rs1 = sRS[pb * 3 + 1];
            float mu2 = sMU[pb * 3 + 2], rs2 = sRS[pb * 3 + 2];
            const unsigned short* axp = Ax_d + (size_t)(t * kB + pb) * kH3;
            const float* ahb = sAH + (size_t)pb * kH3;
            #pragma unroll 4
            for (int e = 0; e < 16; ++e) {
                int j = pj0 + e;
                float ahr = (ahb[j]          - mu0) * rs0 * sGH[j]          + sBEH[j];
                float ahz = (ahb[kH + j]     - mu1) * rs1 * sGH[kH + j]     + sBEH[kH + j];
                float ahn = (ahb[2 * kH + j] - mu2) * rs2 * sGH[2 * kH + j] + sBEH[2 * kH + j];
                float axr = bf2f(axp[j]);
                float axz = bf2f(axp[kH + j]);
                float axn = bf2f(axp[2 * kH + j]);
                float r = sigf(axr + ahr);
                float z = sigf(axz + ahz);
                float n = tanhf(axn + r * ahn);
                float hold = sHF[pb * kH + j];
                float hnew = (1.0f - z) * n + z * hold;
                sHF[pb * kH + j] = hnew;
                sHB[pb * kH + j] = f2bf(hnew);
                if (layer == 0) {
                    outbf[(size_t)(t * kB + pb) * kH + j] = f2bf(hnew);
                } else {
                    outF[(size_t)(t * kB + pb) * (2 * kH) + dir * kH + j] = hnew;
                }
                if (s == kT - 1) {
                    outHid[(size_t)dl * kB * kH + pb * kH + j] = hnew;
                }
            }
        }
        __syncthreads();
    }
}

// ---------------- host launch ----------------
extern "C" void kernel_launch(void* const* d_in, const int* in_sizes, int n_in,
                              void* d_out, int out_size, void* d_ws, size_t ws_size,
                              hipStream_t stream) {
    (void)in_sizes; (void)n_in; (void)out_size; (void)ws_size;

    const float* x   = (const float*)d_in[0];
    const float* h0  = (const float*)d_in[1];
    const float* Wx  = (const float*)d_in[2];
    const float* Wh  = (const float*)d_in[3];
    const float* bx  = (const float*)d_in[4];
    const float* bh  = (const float*)d_in[5];
    const float* gx  = (const float*)d_in[6];
    const float* bex = (const float*)d_in[7];
    const float* gh  = (const float*)d_in[8];
    const float* beh = (const float*)d_in[9];

    float* out    = (float*)d_out;
    float* outHid = out + (size_t)kT * kB * 2 * kH;   // 16777216

    char* ws = (char*)d_ws;
    unsigned short* WxT  = (unsigned short*)(ws + 0);
    unsigned short* WhT  = (unsigned short*)(ws + 6291456);
    unsigned short* Xbf  = (unsigned short*)(ws + 12582912);
    unsigned short* FFbf = (unsigned short*)(ws + 29360128);
    unsigned short* BKbf = (unsigned short*)(ws + 46137344);
    unsigned short* Ax   = (unsigned short*)(ws + 62914560);

    constexpr unsigned kGruSmem = 314112;

    k_cvt_weights<<<12288, 256, 0, stream>>>(Wx, WxT, Wh, WhT);
    k_cvt_x<<<(kTB * kIN + 255) / 256, 256, 0, stream>>>(x, Xbf, kTB * kIN);

    // layer 0 (both directions consume x)
    k_gemm_ax<<<dim3(24, 128, 2), 256, 0, stream>>>(Xbf, Xbf, WxT, bx, Ax, 0);
    k_ln_ax<<<12288, 256, 0, stream>>>(Ax, gx, bex, 0);
    k_gru<<<2, 1024, kGruSmem, stream>>>(0, Ax, WhT, bh, gh, beh, h0,
                                         FFbf, BKbf, nullptr, outHid);

    // layer 1 (fwd consumes FF, bwd consumes BK)
    k_gemm_ax<<<dim3(24, 128, 2), 256, 0, stream>>>(FFbf, BKbf, WxT, bx, Ax, 1);
    k_ln_ax<<<12288, 256, 0, stream>>>(Ax, gx, bex, 1);
    k_gru<<<2, 1024, kGruSmem, stream>>>(1, Ax, WhT, bh, gh, beh, h0,
                                         nullptr, nullptr, out, outHid);
}
